// AFD_35948876268107
// MI455X (gfx1250) — matrix operand order repaired and checked by reading, not compiled, not run on hardware
//
#include <hip/hip_runtime.h>
#include <hip/hip_bf16.h>
#include <math.h>

// ---------------------------------------------------------------------------
// MI455X (gfx1250) implementation.
// Dominant work: two [32768,768]x[768,128] GEMMs -> v_wmma_f32_16x16x32_bf16
// (wave32 XDL path). Input tile staging global->LDS is done with the Tensor
// Data Mover (tensor_load_to_lds + s_wait_tensorcnt): the 16-row tile is one
// contiguous 49152B block, ideal for a single 1-D TDM descriptor per block.
// Everything else is bandwidth-bound streaming. All reductions are
// fixed-order (deterministic). Workspace requirement: ~34 MB.
// ---------------------------------------------------------------------------

typedef __attribute__((ext_vector_type(16))) __bf16 v16bf;
typedef __attribute__((ext_vector_type(8)))  float  v8f;
typedef unsigned int u32x4 __attribute__((ext_vector_type(4)));
typedef int          i32x8 __attribute__((ext_vector_type(8)));
typedef int          i32x4 __attribute__((ext_vector_type(4)));

#define D_DIM   768
#define QK_DIM  128
#define L_DIM   1024
#define S_DIM   4
#define B_DIM   8
#define NROWS   (B_DIM * S_DIM * L_DIM)   // 32768
#define KC_CNT  (D_DIM / 32)              // 24 k-chunks of 32
#define AS_STRIDE 776                     // 768 + 8 halves pad -> stride%64dw==4 (bank rotate)

static __device__ __forceinline__ unsigned short f32_to_bf16_rne(float f) {
    union { float f; unsigned int u; } v; v.f = f;
    unsigned int u = v.u;
    u += 0x7FFFu + ((u >> 16) & 1u);      // round to nearest even
    return (unsigned short)(u >> 16);
}

union FragU { uint4 u[2]; v16bf v; };

// ---------------------------------------------------------------------------
// TDM: 1-D contiguous copy of n_elems (4-byte units, < 65536) from global
// memory to LDS. Descriptor layout per CDNA5 ISA ch.8 (D# groups 0/1).
// ---------------------------------------------------------------------------
static __device__ __forceinline__ void
tdm_load_1d_to_lds(unsigned lds_byte_addr, unsigned long long gaddr, unsigned n_elems) {
    u32x4 g0;
    g0[0] = 1u;                                            // count=1, user desc
    g0[1] = lds_byte_addr;                                 // lds_addr
    g0[2] = (unsigned)(gaddr & 0xFFFFFFFFull);             // global_addr[31:0]
    g0[3] = (unsigned)((gaddr >> 32) & 0x01FFFFFFull)      // global_addr[56:32]
          | (2u << 30);                                    // type = 2 ("image")
    i32x8 g1;
    g1[0] = (int)(2u << 16);                               // data_size=2 (4B), mask=0
    g1[1] = (int)((n_elems & 0xFFFFu) << 16);              // tensor_dim0[15:0]
    g1[2] = (int)((n_elems >> 16) | (1u << 16));           // tensor_dim0[31:16] | tensor_dim1=1
    g1[3] = (int)((n_elems & 0xFFFFu) << 16);              // tile_dim0 = n_elems
    g1[4] = 1;                                             // tile_dim1=1, tile_dim2=0
    g1[5] = (int)n_elems;                                  // tensor_dim0_stride[31:0]
    g1[6] = 0;                                             // stride hi / dim1_stride
    g1[7] = 0;
    i32x4 g2 = {0, 0, 0, 0};
    i32x4 g3 = {0, 0, 0, 0};
#if __clang_major__ >= 23
    i32x8 g4 = {0, 0, 0, 0, 0, 0, 0, 0};
    __builtin_amdgcn_tensor_load_to_lds(g0, g1, g2, g3, g4, 0);
#else
    __builtin_amdgcn_tensor_load_to_lds(g0, g1, g2, g3, 0);
#endif
}

// ---------------------------------------------------------------------------
// Kernel 0: pack W [768][128] f32 -> bf16 in WMMA B-fragment order.
// Pack layout: P[ntile(8)][kc(24)][lane(32)][h(16)] with
//   K = kc*32 + (lane>=16)*16 + h,  N = ntile*16 + (lane&15)
// so each lane's per-kc fragment is 32 contiguous bytes (2 x b128 loads).
// ---------------------------------------------------------------------------
__global__ void __launch_bounds__(256)
wpack_kernel(const float* __restrict__ Wq, const float* __restrict__ Wk,
             unsigned short* __restrict__ wpq, unsigned short* __restrict__ wpk) {
    const float* W = blockIdx.y ? Wk : Wq;
    unsigned short* P = blockIdx.y ? wpk : wpq;
    int idx = blockIdx.x * 256 + threadIdx.x;      // 0 .. 98303
    int h     = idx & 15;
    int lane  = (idx >> 4) & 31;
    int rest  = idx >> 9;
    int kc    = rest % KC_CNT;
    int ntile = rest / KC_CNT;                     // 0..7
    int K = kc * 32 + (lane >> 4) * 16 + h;
    int n = ntile * 16 + (lane & 15);
    P[idx] = f32_to_bf16_rne(W[(size_t)K * QK_DIM + n]);
}

// ---------------------------------------------------------------------------
// Kernel 1: LayerNorm + projection GEMM (WMMA bf16).
// grid.x = 2048 M-tiles of 16 rows, grid.y = {0:feat_t->q, 1:feat_s->k}.
// Block = 256 threads = 8 waves; wave w owns output N-tile w (16 cols).
// Stage A uses the Tensor Data Mover (one descriptor per block).
// ---------------------------------------------------------------------------
__global__ void __launch_bounds__(256)
lnproj_kernel(const float* __restrict__ feat_t, const float* __restrict__ feat_s,
              const float* __restrict__ gq, const float* __restrict__ betaq,
              const unsigned short* __restrict__ wpq, const float* __restrict__ bq,
              const float* __restrict__ gk, const float* __restrict__ betak,
              const unsigned short* __restrict__ wpk, const float* __restrict__ bk,
              float* __restrict__ qbuf, float* __restrict__ kbuf) {
    __shared__ __align__(16) float          F[16 * D_DIM];     // f32 tile (contiguous)
    __shared__ __align__(16) unsigned short A[16 * AS_STRIDE]; // bf16 LN'd A tile

    const int sel = blockIdx.y;
    const float* src            = sel ? feat_s : feat_t;
    const float* gam            = sel ? gk     : gq;
    const float* bet            = sel ? betak  : betaq;
    const unsigned short* wpack = sel ? wpk    : wpq;
    const float* bias           = sel ? bk     : bq;
    float*       dst            = sel ? kbuf   : qbuf;

    const int t    = threadIdx.x;
    const int lane = t & 31;
    const int w    = t >> 5;                       // wave 0..7
    const size_t row_base = (size_t)blockIdx.x * 16;

    // Stage A: TDM DMA of the contiguous 16x768 f32 tile (49152 B) into LDS.
    // Issued by wave 0 only (TDM ignores EXEC; one instruction per wave).
    if (t == 0) {
        // Low 32 bits of a generic pointer to LDS == LDS byte offset (ISA 10.2).
        unsigned lds_addr = (unsigned)(unsigned long long)(const void*)F;
        unsigned long long gaddr =
            (unsigned long long)(const void*)(src + row_base * D_DIM);
        tdm_load_1d_to_lds(lds_addr, gaddr, 16 * D_DIM);   // 12288 dwords
        __builtin_amdgcn_s_wait_tensorcnt(0);
    }
    __syncthreads();

    // Stage B: per-row LayerNorm (wave w handles rows 2w, 2w+1), write bf16.
    for (int rr = 0; rr < 2; ++rr) {
        int r = w * 2 + rr;
        float s = 0.f, s2 = 0.f;
        for (int c = lane; c < D_DIM; c += 32) {
            float x = F[r * D_DIM + c];
            s += x; s2 += x * x;
        }
        for (int m = 16; m > 0; m >>= 1) {
            s  += __shfl_xor(s,  m, 32);
            s2 += __shfl_xor(s2, m, 32);
        }
        float mu   = s * (1.f / D_DIM);
        float var  = s2 * (1.f / D_DIM) - mu * mu;  // biased, like torch LN
        float rstd = rsqrtf(var + 1e-5f);
        for (int c = lane; c < D_DIM; c += 32) {
            float y = (F[r * D_DIM + c] - mu) * rstd * gam[c] + bet[c];
            A[r * AS_STRIDE + c] = f32_to_bf16_rne(y);
        }
    }
    __syncthreads();

    // Stage C: 16x16 output tile per wave, K = 768 in 24 chunks of 32.
    const int m  = lane & 15;       // A row / C col index within tile
    const int hi = lane >> 4;
    const int n  = w * 16 + m;      // global output column
    float bv = bias[n];
    v8f acc = { bv, bv, bv, bv, bv, bv, bv, bv };

    const uint4* bpack = (const uint4*)wpack;       // 8 halves per uint4
    for (int kc = 0; kc < KC_CNT; ++kc) {
        // A fragment (contiguous k order, matching B pack): lane m, halves
        // [kb..kb+15], kb = kc*32 + hi*16.
        int kb = kc * 32 + hi * 16;
        FragU fa, fb;
        fa.u[0] = *(const uint4*)&A[m * AS_STRIDE + kb];
        fa.u[1] = *(const uint4*)&A[m * AS_STRIDE + kb + 8];
        // B fragment: pre-swizzled, 32 contiguous bytes per lane.
        size_t bo = ((size_t)(w * KC_CNT + kc) * 32 + lane) * 2;
        fb.u[0] = bpack[bo];
        fb.u[1] = bpack[bo + 1];
        acc = __builtin_amdgcn_wmma_f32_16x16x32_bf16(
            false, fa.v, false, fb.v, (short)0, acc, false, false);
    }

    // C/D layout: VGPR r -> row (r + 8*hi), col n.
#pragma unroll
    for (int r = 0; r < 8; ++r) {
        size_t row = row_base + r + hi * 8;
        dst[row * QK_DIM + n] = acc[r];
    }
}

// ---------------------------------------------------------------------------
// Kernel 2: logits[b,i,j] = <q[b,i,:,:], k[b,j,:,:]> (contiguous 131072 dot).
// One block per (b,i,j); fixed-order tree reduction (deterministic).
// ---------------------------------------------------------------------------
__global__ void __launch_bounds__(256)
logits_kernel(const float* __restrict__ qbuf, const float* __restrict__ kbuf,
              float* __restrict__ logits) {
    __shared__ float red[256];
    int bij = blockIdx.x;                 // 0..127
    int b = bij >> 4, i = (bij >> 2) & 3, j = bij & 3;
    const float* qp = qbuf + (size_t)(b * 4 + i) * L_DIM * QK_DIM;
    const float* kp = kbuf + (size_t)(b * 4 + j) * L_DIM * QK_DIM;
    float acc = 0.f;
    for (int e = threadIdx.x; e < L_DIM * QK_DIM; e += 256)
        acc += qp[e] * kp[e];
    red[threadIdx.x] = acc; __syncthreads();
    for (int s = 128; s > 0; s >>= 1) {
        if (threadIdx.x < s) red[threadIdx.x] += red[threadIdx.x + s];
        __syncthreads();
    }
    if (threadIdx.x == 0) logits[bij] = red[0];
}

// ---------------------------------------------------------------------------
// Kernel 3: pe + softmax (recomputed per block, trivial) then streaming
// fused = atts . feat_s, accumulate squared error vs feat_t.
// grid.x = 8 (b) * 256 (chunks) = 2048 blocks; per-block partial sums.
// ---------------------------------------------------------------------------
__global__ void __launch_bounds__(256)
fuse_mse_kernel(const float* __restrict__ feat_s, const float* __restrict__ feat_t,
                const float* __restrict__ logits,
                const float* __restrict__ p_t, const float* __restrict__ p_s,
                float* __restrict__ partials) {
    __shared__ float sl[16];
    __shared__ float att[16];
    __shared__ float red[256];
    const int blk = blockIdx.x;
    const int b = blk >> 8;
    const int chunk = blk & 255;
    const int t = threadIdx.x;

    if (t < 16) {
        int i = t >> 2, j = t & 3;
        float pe = 0.f;
        for (int d = 0; d < QK_DIM; ++d) pe += p_t[i * QK_DIM + d] * p_s[j * QK_DIM + d];
        sl[t] = (logits[b * 16 + t] + pe) * 0.08838834764831845f;  // 1/sqrt(128)
    }
    __syncthreads();
    if (t < 4) {
        int i = t;
        float mx = sl[i * 4];
        for (int j = 1; j < 4; ++j) mx = fmaxf(mx, sl[i * 4 + j]);
        float e[4], s = 0.f;
        for (int j = 0; j < 4; ++j) { e[j] = expf(sl[i * 4 + j] - mx); s += e[j]; }
        float inv = 1.f / s;
        for (int j = 0; j < 4; ++j) att[i * 4 + j] = e[j] * inv;
    }
    __syncthreads();

    const size_t LD = (size_t)L_DIM * D_DIM;       // 786432 per (b, stage)
    const float* fs = feat_s + (size_t)b * 4 * LD;
    const float* ft = feat_t + (size_t)b * 4 * LD;
    float acc = 0.f;
    // 786432 positions per b = 256 chunks * 3072; 3072 = 12 iters * 256 thr.
    for (int it = 0; it < 12; ++it) {
        size_t p = (size_t)chunk * 3072 + (size_t)it * 256 + t;
        float s0 = fs[0 * LD + p], s1 = fs[1 * LD + p];
        float s2 = fs[2 * LD + p], s3 = fs[3 * LD + p];
#pragma unroll
        for (int i = 0; i < 4; ++i) {
            float fu = att[i * 4 + 0] * s0 + att[i * 4 + 1] * s1 +
                       att[i * 4 + 2] * s2 + att[i * 4 + 3] * s3;
            float dd = fu - ft[i * LD + p];
            acc += dd * dd;
        }
    }
    red[t] = acc; __syncthreads();
    for (int s = 128; s > 0; s >>= 1) {
        if (t < s) red[t] += red[t + s];
        __syncthreads();
    }
    if (t == 0) partials[blk] = red[0];
}

// ---------------------------------------------------------------------------
// Kernel 4: final deterministic reduction; out = sum * 1/(B*L*D).
// ---------------------------------------------------------------------------
__global__ void __launch_bounds__(256)
final_reduce_kernel(const float* __restrict__ partials, float* __restrict__ out) {
    __shared__ float red[256];
    float acc = 0.f;
    for (int i = threadIdx.x; i < 2048; i += 256) acc += partials[i];
    red[threadIdx.x] = acc; __syncthreads();
    for (int s = 128; s > 0; s >>= 1) {
        if (threadIdx.x < s) red[threadIdx.x] += red[threadIdx.x + s];
        __syncthreads();
    }
    if (threadIdx.x == 0) out[0] = red[0] * (1.0f / 6291456.0f);  // 1/(8*1024*768)
}

// ---------------------------------------------------------------------------
extern "C" void kernel_launch(void* const* d_in, const int* in_sizes, int n_in,
                              void* d_out, int out_size, void* d_ws, size_t ws_size,
                              hipStream_t stream) {
    (void)in_sizes; (void)n_in; (void)out_size; (void)ws_size;
    const float* feat_s = (const float*)d_in[0];
    const float* feat_t = (const float*)d_in[1];
    const float* gq     = (const float*)d_in[2];
    const float* betaq  = (const float*)d_in[3];
    const float* Wq     = (const float*)d_in[4];
    const float* bq     = (const float*)d_in[5];
    const float* gk     = (const float*)d_in[6];
    const float* betak  = (const float*)d_in[7];
    const float* Wk     = (const float*)d_in[8];
    const float* bk     = (const float*)d_in[9];
    const float* p_t    = (const float*)d_in[10];
    const float* p_s    = (const float*)d_in[11];
    float* out = (float*)d_out;

    // Workspace layout (bytes). Total ~33.96 MB.
    char* ws = (char*)d_ws;
    unsigned short* wpq = (unsigned short*)(ws + 0);                 // 196608 B
    unsigned short* wpk = (unsigned short*)(ws + 196608);            // 196608 B
    float* qbuf    = (float*)(ws + 393216);                          // 16.78 MB
    float* kbuf    = (float*)(ws + 393216 + 16777216);               // 16.78 MB
    float* logits  = (float*)(ws + 393216 + 2 * 16777216);           // 512 B
    float* partials= (float*)(ws + 393216 + 2 * 16777216 + 512);     // 8192 B

    wpack_kernel<<<dim3(384, 2), 256, 0, stream>>>(Wq, Wk, wpq, wpk);
    lnproj_kernel<<<dim3(NROWS / 16, 2), 256, 0, stream>>>(
        feat_t, feat_s, gq, betaq, wpq, bq, gk, betak, wpk, bk, qbuf, kbuf);
    logits_kernel<<<128, 256, 0, stream>>>(qbuf, kbuf, logits);
    fuse_mse_kernel<<<2048, 256, 0, stream>>>(feat_s, feat_t, logits, p_t, p_s, partials);
    final_reduce_kernel<<<1, 256, 0, stream>>>(partials, out);
}
